// GRUdecoder_57217554317559
// MI455X (gfx1250) — matrix operand order repaired and checked by reading, not compiled, not run on hardware
//
#include <hip/hip_runtime.h>
#include <hip/hip_bf16.h>

typedef __attribute__((ext_vector_type(16))) __bf16 v16bf;
typedef __attribute__((ext_vector_type(8)))  float  v8f;

// ---------------- helpers ----------------

static __device__ inline unsigned short f2bf(float f) {
    unsigned u = __float_as_uint(f);
    u += 0x7FFFu + ((u >> 16) & 1u);        // round-to-nearest-even
    return (unsigned short)(u >> 16);
}

static __device__ inline float sigmoidf_(float x) {
    return 1.0f / (1.0f + __expf(-x));
}

union FragU { uint4 u[2]; v16bf v; };

// A fragment: 16x32 bf16, row-major source [rows, ld].
// CDNA5 layout: lane<16 holds M=lane, K = {kb..kb+7, 16+kb..16+kb+7}, kb=0; lane>=16 kb=8.
static __device__ inline v16bf load_a(const unsigned short* __restrict__ p,
                                      int ld, int row0, int k0, int lane) {
    const int r  = row0 + (lane & 15);
    const int kb = (lane & 16) ? 8 : 0;
    const unsigned short* base = p + (size_t)r * ld + k0 + kb;
    FragU t;
    t.u[0] = *reinterpret_cast<const uint4*>(base);
    t.u[1] = *reinterpret_cast<const uint4*>(base + 16);
    return t.v;
}

// B fragment: 32x16 bf16 = W^T tile; W stored row-major [Ntot, ld] so B[k][n] = W[n][k].
// Lane = N column; lane<16 holds K={0..7,16..23}, lane>=16 K={8..15,24..31} (2x b128).
static __device__ inline v16bf load_b(const unsigned short* __restrict__ p,
                                      int ld, int n0, int k0, int lane) {
    const int n  = n0 + (lane & 15);
    const int kb = (lane & 16) ? 8 : 0;
    const unsigned short* base = p + (size_t)n * ld + k0 + kb;
    FragU t;
    t.u[0] = *reinterpret_cast<const uint4*>(base);
    t.u[1] = *reinterpret_cast<const uint4*>(base + 16);
    return t.v;
}

#define WMMA_BF16(a, b, c) \
    __builtin_amdgcn_wmma_f32_16x16x32_bf16(false, (a), false, (b), (short)0, (c), false, false)

// ---------------- conversion kernel ----------------

__global__ void cvt_f32_bf16(const float* __restrict__ src,
                             unsigned short* __restrict__ dst, long n) {
    long i = (long)blockIdx.x * blockDim.x + threadIdx.x;
    const long stride = (long)gridDim.x * blockDim.x;
    for (; i < n; i += stride) dst[i] = f2bf(src[i]);
}

// ---------------- GRU gate kernel ----------------
// Each wave: 32x16 tile of h_new. r/z accumulators shared across gi and gh GEMMs.

__global__ __launch_bounds__(256) void gru_gates(
    const unsigned short* __restrict__ xbf,   // (B, I)
    const unsigned short* __restrict__ hbf,   // (B, H)
    const unsigned short* __restrict__ Wih,   // (3H, I)
    const unsigned short* __restrict__ Whh,   // (3H, H)
    const float* __restrict__ bih,            // (3H)
    const float* __restrict__ bhh,            // (3H)
    const float* __restrict__ hprev,          // (B, H) f32
    float* __restrict__ hnew,                 // (B, H) f32
    unsigned short* __restrict__ hnewbf,      // (B, H) bf16
    int I, int H)
{
    const int lane = threadIdx.x & 31;
    const int wave = threadIdx.x >> 5;
    const int mrow = blockIdx.x * 32;
    const int ncol = blockIdx.y * 128 + wave * 16;

    v8f ar0 = {}, ar1 = {};   // r gate (gi + gh joint)
    v8f az0 = {}, az1 = {};   // z gate (gi + gh joint)
    v8f ai0 = {}, ai1 = {};   // i_n part
    v8f ah0 = {}, ah1 = {};   // h_n part

    for (int k0 = 0; k0 < I; k0 += 32) {
        v16bf a0 = load_a(xbf, I, mrow,      k0, lane);
        v16bf a1 = load_a(xbf, I, mrow + 16, k0, lane);
        v16bf br = load_b(Wih, I, ncol,          k0, lane);
        v16bf bz = load_b(Wih, I, H + ncol,      k0, lane);
        v16bf bn = load_b(Wih, I, 2 * H + ncol,  k0, lane);
        ar0 = WMMA_BF16(a0, br, ar0);  ar1 = WMMA_BF16(a1, br, ar1);
        az0 = WMMA_BF16(a0, bz, az0);  az1 = WMMA_BF16(a1, bz, az1);
        ai0 = WMMA_BF16(a0, bn, ai0);  ai1 = WMMA_BF16(a1, bn, ai1);
    }
    for (int k0 = 0; k0 < H; k0 += 32) {
        v16bf a0 = load_a(hbf, H, mrow,      k0, lane);
        v16bf a1 = load_a(hbf, H, mrow + 16, k0, lane);
        v16bf br = load_b(Whh, H, ncol,          k0, lane);
        v16bf bz = load_b(Whh, H, H + ncol,      k0, lane);
        v16bf bn = load_b(Whh, H, 2 * H + ncol,  k0, lane);
        ar0 = WMMA_BF16(a0, br, ar0);  ar1 = WMMA_BF16(a1, br, ar1);
        az0 = WMMA_BF16(a0, bz, az0);  az1 = WMMA_BF16(a1, bz, az1);
        ah0 = WMMA_BF16(a0, bn, ah0);  ah1 = WMMA_BF16(a1, bn, ah1);
    }

    // epilogue: gates + blend. C/D layout: lane<16 -> N=lane, M=vgpr; lane>=16 -> M=vgpr+8.
    const int n = ncol + (lane & 15);
    const float brz = bih[n] + bhh[n];
    const float bzz = bih[H + n] + bhh[H + n];
    const float bin = bih[2 * H + n];
    const float bhn = bhh[2 * H + n];
    const int mofs = (lane & 16) ? 8 : 0;

    #pragma unroll
    for (int mt = 0; mt < 2; ++mt) {
        const v8f& vr = mt ? ar1 : ar0;
        const v8f& vz = mt ? az1 : az0;
        const v8f& vi = mt ? ai1 : ai0;
        const v8f& vh = mt ? ah1 : ah0;
        #pragma unroll
        for (int rr = 0; rr < 8; ++rr) {
            const int m = mrow + mt * 16 + mofs + rr;
            const float rg = sigmoidf_(vr[rr] + brz);
            const float zg = sigmoidf_(vz[rr] + bzz);
            const float ng = tanhf(vi[rr] + bin + rg * (vh[rr] + bhn));
            const size_t idx = (size_t)m * H + n;
            const float hv = (1.0f - zg) * ng + zg * hprev[idx];
            hnew[idx]   = hv;
            hnewbf[idx] = f2bf(hv);
        }
    }
}

// ---------------- FC kernel: x_new = h_new @ W_fc^T + b_fc ----------------

__global__ __launch_bounds__(256) void gru_fc(
    const unsigned short* __restrict__ hbf,   // (B, H)
    const unsigned short* __restrict__ Wfc,   // (I, H)
    const float* __restrict__ bfc,            // (I)
    float* __restrict__ out,                  // (B, T, I), write slot T-1-t
    unsigned short* __restrict__ xbf,         // (B, I) next-step x
    int I, int H, int T, int t)
{
    const int lane = threadIdx.x & 31;
    const int wave = threadIdx.x >> 5;
    const int mrow = blockIdx.x * 32;
    const int ncol = blockIdx.y * 128 + wave * 16;

    v8f acc0 = {}, acc1 = {};
    for (int k0 = 0; k0 < H; k0 += 32) {
        v16bf a0 = load_a(hbf, H, mrow,      k0, lane);
        v16bf a1 = load_a(hbf, H, mrow + 16, k0, lane);
        v16bf b  = load_b(Wfc, H, ncol, k0, lane);
        acc0 = WMMA_BF16(a0, b, acc0);
        acc1 = WMMA_BF16(a1, b, acc1);
    }

    const int n = ncol + (lane & 15);
    const float bias = bfc[n];
    const int mofs = (lane & 16) ? 8 : 0;
    const int trev = T - 1 - t;

    #pragma unroll
    for (int mt = 0; mt < 2; ++mt) {
        const v8f& va = mt ? acc1 : acc0;
        #pragma unroll
        for (int rr = 0; rr < 8; ++rr) {
            const int m = mrow + mt * 16 + mofs + rr;
            const float xv = va[rr] + bias;
            out[(size_t)m * T * I + (size_t)trev * I + n] = xv;
            xbf[(size_t)m * I + n] = f2bf(xv);
        }
    }
}

// ---------------- host ----------------

extern "C" void kernel_launch(void* const* d_in, const int* in_sizes, int n_in,
                              void* d_out, int out_size, void* d_ws, size_t ws_size,
                              hipStream_t stream) {
    const float* x0   = (const float*)d_in[0];   // (B, I)
    const float* h0   = (const float*)d_in[1];   // (B, H)
    const float* Wih  = (const float*)d_in[2];   // (3H, I)
    const float* Whh  = (const float*)d_in[3];   // (3H, H)
    const float* bih  = (const float*)d_in[4];   // (3H)
    const float* bhh  = (const float*)d_in[5];   // (3H)
    const float* Wfc  = (const float*)d_in[6];   // (I, H)
    const float* bfc  = (const float*)d_in[7];   // (I)
    float* out = (float*)d_out;

    const int I = in_sizes[7];           // b_fc length
    const int H = in_sizes[4] / 3;       // b_ih length / 3
    const int B = in_sizes[0] / I;       // inputs (B, I)
    const int T = out_size / (B * I);

    // workspace carve-out (all sizes are multiples of big powers of 2)
    char* w = (char*)d_ws;
    size_t o = 0;
    auto carve = [&](size_t bytes) -> char* {
        char* p = w + o;
        o = (o + bytes + 255) & ~(size_t)255;
        return p;
    };
    unsigned short* Wih_bf  = (unsigned short*)carve((size_t)3 * H * I * 2);
    unsigned short* Whh_bf  = (unsigned short*)carve((size_t)3 * H * H * 2);
    unsigned short* Wfc_bf  = (unsigned short*)carve((size_t)I * H * 2);
    unsigned short* x_bf[2] = { (unsigned short*)carve((size_t)B * I * 2),
                                (unsigned short*)carve((size_t)B * I * 2) };
    unsigned short* h_bf[2] = { (unsigned short*)carve((size_t)B * H * 2),
                                (unsigned short*)carve((size_t)B * H * 2) };
    float* h_f32[2] = { (float*)carve((size_t)B * H * 4),
                        (float*)carve((size_t)B * H * 4) };
    (void)ws_size; (void)n_in;

    // one-shot bf16 conversions (re-done every call; deterministic)
    const int cvB = 256, cvG = 2048;
    cvt_f32_bf16<<<cvG, cvB, 0, stream>>>(Wih, Wih_bf, (long)3 * H * I);
    cvt_f32_bf16<<<cvG, cvB, 0, stream>>>(Whh, Whh_bf, (long)3 * H * H);
    cvt_f32_bf16<<<cvG, cvB, 0, stream>>>(Wfc, Wfc_bf, (long)I * H);
    cvt_f32_bf16<<<cvG, cvB, 0, stream>>>(x0, x_bf[0], (long)B * I);
    cvt_f32_bf16<<<cvG, cvB, 0, stream>>>(h0, h_bf[0], (long)B * H);
    hipMemcpyAsync(h_f32[0], h0, (size_t)B * H * sizeof(float),
                   hipMemcpyDeviceToDevice, stream);

    dim3 gGate(B / 32, H / 128);   // 8 waves/block, each 32x16 of h_new
    dim3 gFc(B / 32, I / 128);

    for (int t = 0; t < T; ++t) {
        const int cur = t & 1, nxt = cur ^ 1;
        gru_gates<<<gGate, 256, 0, stream>>>(
            x_bf[cur], h_bf[cur], Wih_bf, Whh_bf, bih, bhh,
            h_f32[cur], h_f32[nxt], h_bf[nxt], I, H);
        gru_fc<<<gFc, 256, 0, stream>>>(
            h_bf[nxt], Wfc_bf, bfc, out, x_bf[nxt], I, H, T, t);
    }
}
